// TransformerDecoder_11914239279640
// MI455X (gfx1250) — matrix operand order repaired and checked by reading, not compiled, not run on hardware
//
#include <hip/hip_runtime.h>

typedef unsigned short u16;
typedef __bf16  bf16_t;
typedef bf16_t  v16bf __attribute__((ext_vector_type(16)));
typedef float   v8f   __attribute__((ext_vector_type(8)));
typedef unsigned int v4u __attribute__((ext_vector_type(4)));
typedef int v4i __attribute__((ext_vector_type(4)));

// Problem constants (TGT == SRC == 4096, P == 1024, D_FF == 4096)
constexpr int kSEQ = 4096;
constexpr int kP   = 1024;
constexpr int kFF  = 4096;

// GEMM tiling: 128x128 block tile, K-chunk 64 (2 WMMA k-steps per LDS tile)
constexpr int BM = 128, BN = 128, BK = 64;
constexpr int LDT = BK + 8;   // LDS row stride (144B, 16B-aligned)

#if __has_builtin(__builtin_amdgcn_global_load_async_to_lds_b128)
#define USE_ASYNC_COPY 1
#else
#define USE_ASYNC_COPY 0
#endif

__device__ __forceinline__ u16 f32_to_bf16(float f) {
  unsigned int u = __float_as_uint(f);
  u += 0x7fffu + ((u >> 16) & 1u);   // round-to-nearest-even
  return (u16)(u >> 16);
}

__device__ __forceinline__ v16bf ld_frag(const u16* p0, const u16* p1) {
  union { v4u q[2]; v16bf v; } u;
  u.q[0] = *reinterpret_cast<const v4u*>(p0);
  u.q[1] = *reinterpret_cast<const v4u*>(p1);
  return u.v;
}

// 16B global -> LDS copy: async DMA (ASYNCcnt) on gfx1250, else sync copy.
__device__ __forceinline__ void cp16(const u16* __restrict__ g, u16* l) {
#if USE_ASYNC_COPY
  typedef __attribute__((address_space(1))) v4i gv4i;
  typedef __attribute__((address_space(3))) v4i lv4i;
  __builtin_amdgcn_global_load_async_to_lds_b128(
      (gv4i*)(g), (lv4i*)(l), 0, 0);
#else
  *reinterpret_cast<v4u*>(l) = *reinterpret_cast<const v4u*>(g);
#endif
}

__device__ __forceinline__ void wait_async(bool more_in_flight) {
#if USE_ASYNC_COPY
#if __has_builtin(__builtin_amdgcn_s_wait_asynccnt)
  if (more_in_flight) __builtin_amdgcn_s_wait_asynccnt(8);
  else                __builtin_amdgcn_s_wait_asynccnt(0);
#else
  if (more_in_flight) asm volatile("s_wait_asynccnt 0x8" ::: "memory");
  else                asm volatile("s_wait_asynccnt 0x0" ::: "memory");
#endif
#else
  (void)more_in_flight;
#endif
}

// ---------------------------------------------------------------------------
// fp32 -> bf16 elementwise convert
// ---------------------------------------------------------------------------
__global__ __launch_bounds__(256) void cvt_f32_to_bf16_k(
    const float* __restrict__ in, u16* __restrict__ out, int n) {
  int i = blockIdx.x * 256 + threadIdx.x;
  if (i < n) out[i] = f32_to_bf16(in[i]);
}

// ---------------------------------------------------------------------------
// fp32 [R,C] -> bf16 [C,R] tiled transpose-convert (weights, done once)
// ---------------------------------------------------------------------------
__global__ __launch_bounds__(256) void cvt_transpose_bf16_k(
    const float* __restrict__ in, u16* __restrict__ out, int R, int C) {
  __shared__ u16 tile[32][33];
  const int c0 = blockIdx.x * 32, r0 = blockIdx.y * 32;
  const int tx = threadIdx.x & 31, ty = threadIdx.x >> 5;   // 32 x 8
  #pragma unroll
  for (int i = 0; i < 4; ++i) {
    const int r = ty + i * 8;
    tile[r][tx] = f32_to_bf16(in[(size_t)(r0 + r) * C + c0 + tx]);
  }
  __syncthreads();
  #pragma unroll
  for (int i = 0; i < 4; ++i) {
    const int r = ty + i * 8;   // output row = input column
    out[(size_t)(c0 + r) * R + r0 + tx] = tile[tx][r];
  }
}

// ---------------------------------------------------------------------------
// bf16 WMMA GEMM:  out = act( scale * (A @ Bt^T) + bias )
//   A  : [M,K] bf16 row-major
//   Bt : [N,K] bf16 row-major (pre-transposed weight / K-matrix / Vt)
//   OUTF32   : store fp32 to outF, else bf16 to outH
//   TRANSOUT : store bf16 output transposed as [N,M] (for V -> Vt)
// Double-buffered LDS, async global->LDS staging, 16 WMMAs/wave per K tile.
// ---------------------------------------------------------------------------
template<bool OUTF32, bool BIAS, bool RELU, bool TRANSOUT>
__global__ __launch_bounds__(256) void gemm_wmma(
    const u16* __restrict__ A, const u16* __restrict__ Bt,
    const float* __restrict__ bias,
    float* __restrict__ outF, u16* __restrict__ outH,
    int M, int N, int K, float scale)
{
  __shared__ u16 sA[2][BM * LDT];
  __shared__ u16 sB[2][BN * LDT];   // stored [n][k]

  const int tid  = threadIdx.x;
  const int lane = tid & 31;
  const int wave = tid >> 5;
  const int wm   = wave >> 1;    // 0..3 : 32 output rows
  const int wn   = wave & 1;     // 0..1 : 64 output cols
  const int m0   = blockIdx.y * BM;
  const int n0   = blockIdx.x * BN;

  // stage one 128x64 A tile + 128x64 B tile into LDS buffer `buf`
  auto stage = [&](int buf, int k0) {
    #pragma unroll
    for (int t = 0; t < 4; ++t) {
      const int c   = tid + t * 256;   // 0..1023 chunks of 16B
      const int r   = c >> 3;          // 0..127
      const int col = (c & 7) * 8;     // 0..56
      cp16(A  + (size_t)(m0 + r) * K + k0 + col, &sA[buf][r * LDT + col]);
      cp16(Bt + (size_t)(n0 + r) * K + k0 + col, &sB[buf][r * LDT + col]);
    }
  };

  v8f zero;
  #pragma unroll
  for (int r = 0; r < 8; ++r) zero[r] = 0.0f;
  v8f acc[2][4];
  #pragma unroll
  for (int i = 0; i < 2; ++i)
    #pragma unroll
    for (int j = 0; j < 4; ++j) acc[i][j] = zero;

  stage(0, 0);
  const int ntile = K / BK;
  for (int it = 0; it < ntile; ++it) {
    const int  cur  = it & 1;
    const bool more = (it + 1 < ntile);
    if (more) stage(cur ^ 1, (it + 1) * BK);   // prefetch next tile (async)
    wait_async(more);                           // current tile's copies done
    __syncthreads();

    const int l15 = lane & 15;
    const int kbA = (lane >> 4) * 8;   // A frag: lane<16 -> K{0..7,16..23}
    const int kbB = (lane >> 4) * 8;   // B frag: lane<16 -> K{0..7,16..23}
    #pragma unroll
    for (int ks = 0; ks < BK; ks += 32) {
      v16bf af[2];
      #pragma unroll
      for (int i = 0; i < 2; ++i) {
        const u16* pa = &sA[cur][(wm * 32 + i * 16 + l15) * LDT + ks + kbA];
        af[i] = ld_frag(pa, pa + 16);
      }
      #pragma unroll
      for (int j = 0; j < 4; ++j) {
        const u16* pb = &sB[cur][(wn * 64 + j * 16 + l15) * LDT + ks + kbB];
        v16bf bfr = ld_frag(pb, pb + 16);
        #pragma unroll
        for (int i = 0; i < 2; ++i)
          acc[i][j] = __builtin_amdgcn_wmma_f32_16x16x32_bf16(
              false, af[i], false, bfr, (short)0, acc[i][j], false, false);
      }
    }
    __syncthreads();   // all waves done reading buf `cur` before it is reused
  }

  // ---- epilogue: C/D layout -> lane = col, VGPR r = row (+8 hi half) ----
  const int l15 = lane & 15;
  const int mh  = (lane >> 4) * 8;
  #pragma unroll
  for (int j = 0; j < 4; ++j) {
    const int n  = n0 + wn * 64 + j * 16 + l15;
    const float bv = BIAS ? bias[n] : 0.0f;
    #pragma unroll
    for (int i = 0; i < 2; ++i) {
      const int mb = m0 + wm * 32 + i * 16 + mh;
      if (TRANSOUT) {
        // transposed bf16 store: contiguous 8 elements along M per lane
        union { v4u q; u16 h[8]; } t;
        #pragma unroll
        for (int r = 0; r < 8; ++r) {
          float c = acc[i][j][r] * scale + bv;
          if (RELU) c = fmaxf(c, 0.0f);
          t.h[r] = f32_to_bf16(c);
        }
        *reinterpret_cast<v4u*>(outH + (size_t)n * M + mb) = t.q;
      } else {
        #pragma unroll
        for (int r = 0; r < 8; ++r) {
          float c = acc[i][j][r] * scale + bv;
          if (RELU) c = fmaxf(c, 0.0f);
          if (OUTF32) outF[(size_t)(mb + r) * N + n] = c;
          else        outH[(size_t)(mb + r) * N + n] = f32_to_bf16(c);
        }
      }
    }
  }
}

// ---------------------------------------------------------------------------
// Row softmax (optionally causal), fp32 scores in, bf16 probabilities out.
// ---------------------------------------------------------------------------
template<bool CAUSAL>
__global__ __launch_bounds__(256) void softmax_rows_bf16(
    const float* __restrict__ S, u16* __restrict__ A, int ncols)
{
  const int row = blockIdx.x;
  const int lim = CAUSAL ? (row + 1) : ncols;
  const float* s = S + (size_t)row * ncols;
  u16* a = A + (size_t)row * ncols;
  __shared__ float red[8];

  float m = -3.0e38f;
  for (int j = threadIdx.x; j < lim; j += 256) m = fmaxf(m, s[j]);
  #pragma unroll
  for (int off = 16; off; off >>= 1) m = fmaxf(m, __shfl_xor(m, off));
  if ((threadIdx.x & 31) == 0) red[threadIdx.x >> 5] = m;
  __syncthreads();
  float mall = red[0];
  #pragma unroll
  for (int i = 1; i < 8; ++i) mall = fmaxf(mall, red[i]);
  __syncthreads();

  float sum = 0.0f;
  for (int j = threadIdx.x; j < lim; j += 256) sum += __expf(s[j] - mall);
  #pragma unroll
  for (int off = 16; off; off >>= 1) sum += __shfl_xor(sum, off);
  if ((threadIdx.x & 31) == 0) red[threadIdx.x >> 5] = sum;
  __syncthreads();
  float tot = 0.0f;
  #pragma unroll
  for (int i = 0; i < 8; ++i) tot += red[i];
  const float inv = 1.0f / tot;

  for (int j = threadIdx.x; j < ncols; j += 256) {
    float p = (j < lim) ? __expf(s[j] - mall) * inv : 0.0f;
    a[j] = f32_to_bf16(p);
  }
}

// ---------------------------------------------------------------------------
// Fused residual add + LayerNorm over D=1024; fp32 out (+ optional bf16).
// ---------------------------------------------------------------------------
__global__ __launch_bounds__(256) void add_ln_1024(
    const float* __restrict__ X, const float* __restrict__ T,
    const float* __restrict__ g, const float* __restrict__ b,
    float* __restrict__ outF, u16* __restrict__ outH)
{
  const int row = blockIdx.x;
  const size_t base = (size_t)row * kP;
  __shared__ float red[8];

  float v[4];
  float s = 0.0f;
  #pragma unroll
  for (int i = 0; i < 4; ++i) {
    const int c = threadIdx.x + i * 256;
    v[i] = X[base + c] + T[base + c];
    s += v[i];
  }
  #pragma unroll
  for (int off = 16; off; off >>= 1) s += __shfl_xor(s, off);
  if ((threadIdx.x & 31) == 0) red[threadIdx.x >> 5] = s;
  __syncthreads();
  float tot = 0.0f;
  #pragma unroll
  for (int i = 0; i < 8; ++i) tot += red[i];
  const float mu = tot * (1.0f / kP);
  __syncthreads();

  float s2 = 0.0f;
  #pragma unroll
  for (int i = 0; i < 4; ++i) { const float d = v[i] - mu; s2 += d * d; }
  #pragma unroll
  for (int off = 16; off; off >>= 1) s2 += __shfl_xor(s2, off);
  if ((threadIdx.x & 31) == 0) red[threadIdx.x >> 5] = s2;
  __syncthreads();
  float tot2 = 0.0f;
  #pragma unroll
  for (int i = 0; i < 8; ++i) tot2 += red[i];
  const float rstd = rsqrtf(tot2 * (1.0f / kP) + 1e-5f);

  #pragma unroll
  for (int i = 0; i < 4; ++i) {
    const int c = threadIdx.x + i * 256;
    const float y = (v[i] - mu) * rstd * g[c] + b[c];
    outF[base + c] = y;
    if (outH) outH[base + c] = f32_to_bf16(y);
  }
}

// ---------------------------------------------------------------------------
// Host: full decoder layer pipeline, all on `stream`, scratch from d_ws.
// ---------------------------------------------------------------------------
extern "C" void kernel_launch(void* const* d_in, const int* in_sizes, int n_in,
                              void* d_out, int out_size, void* d_ws, size_t ws_size,
                              hipStream_t stream)
{
  (void)in_sizes; (void)n_in; (void)out_size; (void)ws_size;
  const float* Y   = (const float*)d_in[0];
  const float* Xp  = (const float*)d_in[1];
  const float* wq1 = (const float*)d_in[2];  const float* bq1 = (const float*)d_in[3];
  const float* wk1 = (const float*)d_in[4];  const float* bk1 = (const float*)d_in[5];
  const float* wv1 = (const float*)d_in[6];  const float* bv1 = (const float*)d_in[7];
  const float* w0  = (const float*)d_in[8];  const float* b0  = (const float*)d_in[9];
  const float* g1  = (const float*)d_in[10]; const float* be1 = (const float*)d_in[11];
  const float* wq2 = (const float*)d_in[12]; const float* bq2 = (const float*)d_in[13];
  const float* wk2 = (const float*)d_in[14]; const float* bk2 = (const float*)d_in[15];
  const float* wv2 = (const float*)d_in[16]; const float* bv2 = (const float*)d_in[17];
  const float* w1  = (const float*)d_in[18]; const float* b1  = (const float*)d_in[19];
  const float* g2  = (const float*)d_in[20]; const float* be2 = (const float*)d_in[21];
  const float* w2  = (const float*)d_in[22]; const float* b2  = (const float*)d_in[23];
  const float* w3  = (const float*)d_in[24]; const float* b3  = (const float*)d_in[25];
  const float* g3  = (const float*)d_in[26]; const float* be3 = (const float*)d_in[27];

  // --- workspace arena ---
  char* ws = (char*)d_ws;
  size_t off = 0;
  auto alloc = [&](size_t bytes) -> void* {
    void* p = ws + off;
    off += (bytes + 255) & ~(size_t)255;
    return p;
  };
  const size_t actB = (size_t)kSEQ * kP * 2;   // bf16 [4096,1024]
  const size_t wB   = (size_t)kP * kP * 2;     // bf16 [1024,1024]

  u16* Yb   = (u16*)alloc(actB);
  u16* Xpb  = (u16*)alloc(actB);
  u16* wq1t = (u16*)alloc(wB);   // all weights stored transposed [N,K]
  u16* wk1t = (u16*)alloc(wB);
  u16* wv1t = (u16*)alloc(wB);
  u16* w0t  = (u16*)alloc(wB);
  u16* wq2t = (u16*)alloc(wB);
  u16* wk2t = (u16*)alloc(wB);
  u16* wv2t = (u16*)alloc(wB);
  u16* w1t  = (u16*)alloc(wB);
  u16* w2t  = (u16*)alloc((size_t)kP * kFF * 2);   // [kFF, kP]
  u16* w3t  = (u16*)alloc((size_t)kFF * kP * 2);   // [kP, kFF]
  u16* Qb   = (u16*)alloc(actB);
  u16* Kb   = (u16*)alloc(actB);                   // [SEQ,P] == [N,K] for scores
  u16* Vt   = (u16*)alloc(actB);                   // [P,SEQ] transposed V
  u16* AVb  = (u16*)alloc(actB);
  float* S  = (float*)alloc((size_t)kSEQ * kSEQ * 4);   // 64MB scores
  u16* Ab   = (u16*)alloc((size_t)kSEQ * kSEQ * 2);     // 32MB probs
  float* T  = (float*)alloc((size_t)kSEQ * kP * 4);
  float* U1f = (float*)alloc((size_t)kSEQ * kP * 4);
  float* U2f = (float*)alloc((size_t)kSEQ * kP * 4);
  u16* U1b  = (u16*)alloc(actB);
  u16* Hb   = (u16*)S;    // reuse 64MB score region for 32MB ReLU activations
  u16* U2b  = U1b;        // U1b dead after Q2 projection

  const dim3 blk(256);
  auto cvt = [&](const float* in, u16* out, size_t n) {
    cvt_f32_to_bf16_k<<<dim3((unsigned)((n + 255) / 256)), blk, 0, stream>>>(in, out, (int)n);
  };
  auto cvtT = [&](const float* in, u16* out, int R, int C) {
    cvt_transpose_bf16_k<<<dim3(C / 32, R / 32), blk, 0, stream>>>(in, out, R, C);
  };
  cvt(Y,  Yb,  (size_t)kSEQ * kP);
  cvt(Xp, Xpb, (size_t)kSEQ * kP);
  cvtT(wq1, wq1t, kP, kP);  cvtT(wk1, wk1t, kP, kP);
  cvtT(wv1, wv1t, kP, kP);  cvtT(w0,  w0t,  kP, kP);
  cvtT(wq2, wq2t, kP, kP);  cvtT(wk2, wk2t, kP, kP);
  cvtT(wv2, wv2t, kP, kP);  cvtT(w1,  w1t,  kP, kP);
  cvtT(w2,  w2t,  kP, kFF); cvtT(w3,  w3t,  kFF, kP);

  const float scl = 0.03125f;                    // 1/sqrt(P)
  const dim3 gP(kP / BN,  kSEQ / BM);            // N = 1024
  const dim3 gS(kSEQ / BN, kSEQ / BM);           // N = 4096 (scores)
  const dim3 gF(kFF / BN, kSEQ / BM);            // N = 4096 (FFN up)

  // template args: <OUTF32, BIAS, RELU, TRANSOUT>
  // ---- masked self-attention ----
  gemm_wmma<false,true ,false,false><<<gP, blk, 0, stream>>>(Yb, wq1t, bq1, nullptr, Qb, kSEQ, kP, kP, 1.0f);
  gemm_wmma<false,true ,false,false><<<gP, blk, 0, stream>>>(Yb, wk1t, bk1, nullptr, Kb, kSEQ, kP, kP, 1.0f);
  gemm_wmma<false,true ,false,true ><<<gP, blk, 0, stream>>>(Yb, wv1t, bv1, nullptr, Vt, kSEQ, kP, kP, 1.0f);
  gemm_wmma<true ,false,false,false><<<gS, blk, 0, stream>>>(Qb, Kb, nullptr, S, nullptr, kSEQ, kSEQ, kP, scl);
  softmax_rows_bf16<true><<<dim3(kSEQ), blk, 0, stream>>>(S, Ab, kSEQ);
  gemm_wmma<false,false,false,false><<<gP, blk, 0, stream>>>(Ab, Vt, nullptr, nullptr, AVb, kSEQ, kP, kSEQ, 1.0f);
  gemm_wmma<true ,true ,false,false><<<gP, blk, 0, stream>>>(AVb, w0t, b0, T, nullptr, kSEQ, kP, kP, 1.0f);
  add_ln_1024<<<dim3(kSEQ), blk, 0, stream>>>(Y, T, g1, be1, U1f, U1b);

  // ---- cross-attention ----
  gemm_wmma<false,true ,false,false><<<gP, blk, 0, stream>>>(U1b, wq2t, bq2, nullptr, Qb, kSEQ, kP, kP, 1.0f);
  gemm_wmma<false,true ,false,false><<<gP, blk, 0, stream>>>(Xpb, wk2t, bk2, nullptr, Kb, kSEQ, kP, kP, 1.0f);
  gemm_wmma<false,true ,false,true ><<<gP, blk, 0, stream>>>(Xpb, wv2t, bv2, nullptr, Vt, kSEQ, kP, kP, 1.0f);
  gemm_wmma<true ,false,false,false><<<gS, blk, 0, stream>>>(Qb, Kb, nullptr, S, nullptr, kSEQ, kSEQ, kP, scl);
  softmax_rows_bf16<false><<<dim3(kSEQ), blk, 0, stream>>>(S, Ab, kSEQ);
  gemm_wmma<false,false,false,false><<<gP, blk, 0, stream>>>(Ab, Vt, nullptr, nullptr, AVb, kSEQ, kP, kSEQ, 1.0f);
  gemm_wmma<true ,true ,false,false><<<gP, blk, 0, stream>>>(AVb, w1t, b1, T, nullptr, kSEQ, kP, kP, 1.0f);
  add_ln_1024<<<dim3(kSEQ), blk, 0, stream>>>(U1f, T, g2, be2, U2f, U2b);

  // ---- position-wise FFN ----
  gemm_wmma<false,true ,true ,false><<<gF, blk, 0, stream>>>(U2b, w2t, b2, nullptr, Hb, kSEQ, kFF, kP, 1.0f);
  gemm_wmma<true ,true ,false,false><<<gP, blk, 0, stream>>>(Hb, w3t, b3, T, nullptr, kSEQ, kP, kFF, 1.0f);
  add_ln_1024<<<dim3(kSEQ), blk, 0, stream>>>(U2f, T, g3, be3, (float*)d_out, nullptr);
}
